// UltraMultiHeadAttention_64939905515570
// MI455X (gfx1250) — matrix operand order repaired and checked by reading, not compiled, not run on hardware
//
#include <hip/hip_runtime.h>
#include <hip/hip_bf16.h>

typedef __bf16 bf16_t;
typedef __attribute__((ext_vector_type(16))) __bf16 v16bf;
typedef __attribute__((ext_vector_type(8)))  __bf16 v8bf;
typedef __attribute__((ext_vector_type(8)))  float  v8f;

#define WMMA_BF16(a, b, c) \
    __builtin_amdgcn_wmma_f32_16x16x32_bf16(false, (a), false, (b), (short)0, (c), false, false)

static __device__ __forceinline__ v8f vzero8f() {
    v8f z;
#pragma unroll
    for (int i = 0; i < 8; ++i) z[i] = 0.0f;
    return z;
}

static __device__ __forceinline__ v16bf cat16(v8bf lo, v8bf hi) {
    v16bf r;
#pragma unroll
    for (int i = 0; i < 8; ++i) { r[i] = lo[i]; r[i + 8] = hi[i]; }
    return r;
}

static __device__ __forceinline__ float half_max(float v) {
#pragma unroll
    for (int off = 1; off < 16; off <<= 1) v = fmaxf(v, __shfl_xor(v, off, 32));
    return v;
}
static __device__ __forceinline__ float half_sum(float v) {
#pragma unroll
    for (int off = 1; off < 16; off <<= 1) v += __shfl_xor(v, off, 32);
    return v;
}

// ---------------------------------------------------------------------------
// Kernel 1: fp32 -> bf16 flat convert
// ---------------------------------------------------------------------------
__global__ __launch_bounds__(256) void f32_to_bf16_vec(
    const float* __restrict__ src, bf16_t* __restrict__ dst, int n)
{
    int i = (blockIdx.x * 256 + threadIdx.x) * 4;
    if (i + 3 < n) {
        float4 v = *(const float4*)(src + i);
        dst[i + 0] = (bf16_t)v.x;
        dst[i + 1] = (bf16_t)v.y;
        dst[i + 2] = (bf16_t)v.z;
        dst[i + 3] = (bf16_t)v.w;
    }
}

// ---------------------------------------------------------------------------
// Kernel 2: transpose fp32 [R][C] -> bf16 [C][R]  (N-major weights for WMMA B)
// ---------------------------------------------------------------------------
__global__ __launch_bounds__(256) void transpose_f32_to_bf16(
    const float* __restrict__ src, bf16_t* __restrict__ dst, int R, int C)
{
    __shared__ float tile[32][33];
    int c0 = blockIdx.x * 32, r0 = blockIdx.y * 32;
    int x = threadIdx.x, y0 = threadIdx.y;
#pragma unroll
    for (int j = y0; j < 32; j += 8)
        tile[j][x] = src[(size_t)(r0 + j) * C + c0 + x];
    __syncthreads();
#pragma unroll
    for (int j = y0; j < 32; j += 8)
        dst[(size_t)(c0 + j) * R + r0 + x] = (bf16_t)tile[x][j];
}

// ---------------------------------------------------------------------------
// Kernel 3/5: bf16 WMMA GEMM,  C[M,N] = A[M,K] * Bt[N,K]^T
//   block tile 128x128, BK=32, 256 threads = 8 waves (2M x 4N), wave = 64x32.
//   MODE 0: store fp32 C directly.
//   MODE 1: QKV epilogue -> scatter bf16 Q[B,NH,S,HD], K[B,NH,S,HD], Vt[B,NH,HD,S]
// ---------------------------------------------------------------------------
template <int MODE>
__global__ __launch_bounds__(256) void gemm_bf16(
    const bf16_t* __restrict__ A, const bf16_t* __restrict__ Bt,
    float* __restrict__ Cout,
    bf16_t* __restrict__ Qo, bf16_t* __restrict__ Ko, bf16_t* __restrict__ Vto,
    int Ndim, int Kdim)
{
    __shared__ bf16_t As[128 * 32];
    __shared__ bf16_t Bs[128 * 32];

    const int tid  = threadIdx.x;
    const int lane = tid & 31;
    const int wid  = tid >> 5;
    const int l16  = lane & 15;
    const int lh   = lane >> 4;
    const int m0   = blockIdx.y * 128;
    const int n0   = blockIdx.x * 128;
    const int wm   = (wid >> 2) * 64;   // wave M offset: 0 or 64
    const int wn   = (wid & 3) * 32;    // wave N offset: 0..96

    v8f acc[4][2];
#pragma unroll
    for (int i = 0; i < 4; ++i)
#pragma unroll
        for (int j = 0; j < 2; ++j) acc[i][j] = vzero8f();

    const int ldrow = tid >> 1;         // 0..127
    const int ldhalf = tid & 1;         // 0..1

    for (int k0 = 0; k0 < Kdim; k0 += 32) {
        // ---- cooperative tile loads (each thread: 32B of A + 32B of B) ----
        {
            const uint4* sa = (const uint4*)(A + (size_t)(m0 + ldrow) * Kdim + k0) + ldhalf * 2;
            uint4* da = (uint4*)(As + ldrow * 32) + ldhalf * 2;
            da[0] = sa[0]; da[1] = sa[1];
            const uint4* sb = (const uint4*)(Bt + (size_t)(n0 + ldrow) * Kdim + k0) + ldhalf * 2;
            uint4* db = (uint4*)(Bs + ldrow * 32) + ldhalf * 2;
            db[0] = sb[0]; db[1] = sb[1];
            if (k0 + 32 < Kdim)
                __builtin_prefetch(A + (size_t)(m0 + ldrow) * Kdim + k0 + 32, 0, 0);
        }
        __syncthreads();

        // ---- fragments + 8 WMMAs per wave ----
        v16bf af[4];
#pragma unroll
        for (int mf = 0; mf < 4; ++mf) {
            const bf16_t* arow = As + (wm + mf * 16 + l16) * 32;
            af[mf] = cat16(*(const v8bf*)(arow + lh * 16),
                           *(const v8bf*)(arow + lh * 16 + 8));
        }
#pragma unroll
        for (int nf = 0; nf < 2; ++nf) {
            const bf16_t* brow = Bs + (wn + nf * 16 + l16) * 32;
            v16bf bfrag = *(const v16bf*)(brow + lh * 16);
#pragma unroll
            for (int mf = 0; mf < 4; ++mf)
                acc[mf][nf] = WMMA_BF16(af[mf], bfrag, acc[mf][nf]);
        }
        __syncthreads();
    }

    // ---- epilogue ----
#pragma unroll
    for (int mf = 0; mf < 4; ++mf) {
#pragma unroll
        for (int nf = 0; nf < 2; ++nf) {
            const int n = n0 + wn + nf * 16 + l16;
            if (MODE == 0) {
#pragma unroll
                for (int r = 0; r < 8; ++r) {
                    const int m = m0 + wm + mf * 16 + lh * 8 + r;
                    Cout[(size_t)m * Ndim + n] = acc[mf][nf][r];
                }
            } else {
                const int which = n >> 11;          // 0=Q 1=K 2=V
                const int head  = (n & 2047) >> 7;
                const int d     = n & 127;
#pragma unroll
                for (int r = 0; r < 8; ++r) {
                    const int m = m0 + wm + mf * 16 + lh * 8 + r;
                    const int b = m >> 11;
                    const int s = m & 2047;
                    const bf16_t v = (bf16_t)acc[mf][nf][r];
                    const size_t bh = (size_t)(b * 16 + head);
                    if (which == 0)      Qo[(bh * 2048 + s) * 128 + d] = v;
                    else if (which == 1) Ko[(bh * 2048 + s) * 128 + d] = v;
                    else                 Vto[(bh * 128 + d) * 2048 + s] = v;
                }
            }
        }
    }
}

// ---------------------------------------------------------------------------
// Kernel 4: flash attention (causal + pad mask), bf16 WMMA, f32 softmax state.
//   grid = (S/128, NH, B), block = 256 (8 waves). Wave owns 16 query rows.
//   Key tiles of 64. Q[B,NH,S,HD], K[B,NH,S,HD], Vt[B,NH,HD,S] -> attn[B,S,H] bf16
// ---------------------------------------------------------------------------
__global__ __launch_bounds__(256) void flash_attn(
    const bf16_t* __restrict__ Q, const bf16_t* __restrict__ Kg,
    const bf16_t* __restrict__ Vt, const int* __restrict__ amask,
    bf16_t* __restrict__ attn)
{
    __shared__ bf16_t Ks[64 * 128];     // [key][d]  = B-layout for Q*K^T
    __shared__ bf16_t Vs[128 * 64];     // [d][key]  = B-layout for P*V
    __shared__ bf16_t Ps[8][16 * 64];   // per-wave P buffer (C->A relayout)

    const int tid = threadIdx.x, lane = tid & 31, wid = tid >> 5;
    const int l16 = lane & 15, lh = lane >> 4;
    const int qt = blockIdx.x, h = blockIdx.y, b = blockIdx.z;

    const size_t bh = (size_t)(b * 16 + h);
    const bf16_t* Qg  = Q  + bh * 2048 * 128;
    const bf16_t* Kgb = Kg + bh * 2048 * 128;
    const bf16_t* Vtb = Vt + bh * 128 * 2048;
    const int* mrow = amask + b * 2048;
    const int qbase = qt * 128 + wid * 16;       // this wave's first query (global)

    // preload Q fragments (A-layout) straight from global
    v16bf qf[4];
    const bf16_t* qrow = Qg + (size_t)(qbase + l16) * 128;
#pragma unroll
    for (int kk = 0; kk < 4; ++kk)
        qf[kk] = cat16(*(const v8bf*)(qrow + kk * 32 + lh * 16),
                       *(const v8bf*)(qrow + kk * 32 + lh * 16 + 8));

    float mst[8], lst[8];
    v8f Of[8];
#pragma unroll
    for (int r = 0; r < 8; ++r) { mst[r] = -1e30f; lst[r] = 0.0f; }
#pragma unroll
    for (int d = 0; d < 8; ++d) Of[d] = vzero8f();

    const float scale = 0.08838834764831845f;    // 1/sqrt(128)
    const int ntiles = qt * 2 + 2;               // causal: keys <= qt*128+127

    for (int t = 0; t < ntiles; ++t) {
        const int n0 = t * 64;
        __syncthreads();   // previous tile fully consumed before overwrite
        {   // K tile: 64 rows x 128 -> contiguous 16KB
            const uint4* src = (const uint4*)(Kgb + (size_t)n0 * 128);
            uint4* dst = (uint4*)Ks;
#pragma unroll
            for (int i = 0; i < 4; ++i) dst[tid * 4 + i] = src[tid * 4 + i];
            // Vt tile: 128 rows x 64 cols from [128][2048]
            const int row = tid >> 1, hf = tid & 1;
            const uint4* vsrc = (const uint4*)(Vtb + (size_t)row * 2048 + n0 + hf * 32);
            uint4* vdst = (uint4*)(Vs + row * 64 + hf * 32);
#pragma unroll
            for (int i = 0; i < 4; ++i) vdst[i] = vsrc[i];
        }
        __syncthreads();

        // ---- scores S = Q * K^T : 16 WMMAs ----
        v8f Sf[4];
#pragma unroll
        for (int nf = 0; nf < 4; ++nf) Sf[nf] = vzero8f();
#pragma unroll
        for (int kk = 0; kk < 4; ++kk) {
#pragma unroll
            for (int nf = 0; nf < 4; ++nf) {
                v16bf bk = *(const v16bf*)(Ks + (nf * 16 + l16) * 128 + kk * 32 + lh * 16);
                Sf[nf] = WMMA_BF16(qf[kk], bk, Sf[nf]);
            }
        }

        // ---- scale + causal + pad mask ----
        const int qrow0 = qbase + lh * 8;
#pragma unroll
        for (int nf = 0; nf < 4; ++nf) {
            const int kidx = n0 + nf * 16 + l16;
            const int mv = mrow[kidx];
#pragma unroll
            for (int r = 0; r < 8; ++r) {
                float v = Sf[nf][r] * scale;
                if (kidx > qrow0 + r || mv == 0) v = -1e9f;
                Sf[nf][r] = v;
            }
        }

        // ---- online softmax ----
        float alpha[8];
#pragma unroll
        for (int r = 0; r < 8; ++r) {
            float tmax = fmaxf(fmaxf(Sf[0][r], Sf[1][r]), fmaxf(Sf[2][r], Sf[3][r]));
            tmax = half_max(tmax);
            const float mn = fmaxf(mst[r], tmax);
            alpha[r] = __expf(mst[r] - mn);
            mst[r] = mn;
            lst[r] *= alpha[r];
        }
#pragma unroll
        for (int d = 0; d < 8; ++d)
#pragma unroll
            for (int r = 0; r < 8; ++r) Of[d][r] *= alpha[r];

        bf16_t* Pw = Ps[wid];
        float rsum[8];
#pragma unroll
        for (int r = 0; r < 8; ++r) rsum[r] = 0.0f;
#pragma unroll
        for (int nf = 0; nf < 4; ++nf) {
#pragma unroll
            for (int r = 0; r < 8; ++r) {
                const float p = __expf(Sf[nf][r] - mst[r]);
                rsum[r] += p;
                Pw[(lh * 8 + r) * 64 + nf * 16 + l16] = (bf16_t)p;
            }
        }
#pragma unroll
        for (int r = 0; r < 8; ++r) lst[r] += half_sum(rsum[r]);

        // ---- O += P * V : 16 WMMAs ----
#pragma unroll
        for (int ks = 0; ks < 2; ++ks) {
            const bf16_t* prow = Pw + l16 * 64 + ks * 32;
            v16bf ap = cat16(*(const v8bf*)(prow + lh * 16),
                             *(const v8bf*)(prow + lh * 16 + 8));
#pragma unroll
            for (int d = 0; d < 8; ++d) {
                v16bf bv = *(const v16bf*)(Vs + (d * 16 + l16) * 64 + ks * 32 + lh * 16);
                Of[d] = WMMA_BF16(ap, bv, Of[d]);
            }
        }
    }

    // ---- finalize: O /= l, write attn[B,S,H] bf16 ----
#pragma unroll
    for (int r = 0; r < 8; ++r) {
        const float inv = 1.0f / lst[r];
        const size_t row = (size_t)(b * 2048 + qbase + lh * 8 + r) * 2048 + h * 128;
#pragma unroll
        for (int d = 0; d < 8; ++d)
            attn[row + d * 16 + l16] = (bf16_t)(Of[d][r] * inv);
    }
}

// ---------------------------------------------------------------------------
// Host launch
// ---------------------------------------------------------------------------
extern "C" void kernel_launch(void* const* d_in, const int* in_sizes, int n_in,
                              void* d_out, int out_size, void* d_ws, size_t ws_size,
                              hipStream_t stream)
{
    (void)in_sizes; (void)n_in; (void)out_size; (void)ws_size;
    const float* X     = (const float*)d_in[0];   // [2,2048,2048]
    const int*   amask = (const int*)  d_in[1];   // [2,2048]
    const float* Wqkv  = (const float*)d_in[2];   // [2048,6144]
    const float* Wout  = (const float*)d_in[3];   // [2048,2048]
    float* out = (float*)d_out;                   // [2,2048,2048] fp32

    const size_t SZ_XBF   = (size_t)4096 * 2048;
    const size_t SZ_WQKVT = (size_t)6144 * 2048;
    const size_t SZ_WOUTT = (size_t)2048 * 2048;
    const size_t SZ_QKV   = (size_t)2 * 16 * 2048 * 128;
    const size_t SZ_ATTN  = (size_t)4096 * 2048;

    bf16_t* p      = (bf16_t*)d_ws;
    bf16_t* Xbf    = p; p += SZ_XBF;
    bf16_t* WqkvT  = p; p += SZ_WQKVT;
    bf16_t* WoutT  = p; p += SZ_WOUTT;
    bf16_t* Qb     = p; p += SZ_QKV;
    bf16_t* Kb     = p; p += SZ_QKV;
    bf16_t* Vtb    = p; p += SZ_QKV;
    bf16_t* attnb  = p; p += SZ_ATTN;

    // 1) hidden_states -> bf16
    f32_to_bf16_vec<<<8192, 256, 0, stream>>>(X, Xbf, 4096 * 2048);
    // 2) weight transposes (N-major bf16)
    transpose_f32_to_bf16<<<dim3(192, 64), dim3(32, 8), 0, stream>>>(Wqkv, WqkvT, 2048, 6144);
    transpose_f32_to_bf16<<<dim3(64, 64),  dim3(32, 8), 0, stream>>>(Wout, WoutT, 2048, 2048);
    // 3) QKV GEMM + scatter (Q,K row-major; V transposed)
    gemm_bf16<1><<<dim3(48, 32), 256, 0, stream>>>(Xbf, WqkvT, nullptr, Qb, Kb, Vtb, 6144, 2048);
    // 4) flash attention
    flash_attn<<<dim3(16, 16, 2), 256, 0, stream>>>(Qb, Kb, Vtb, amask, attnb);
    // 5) output projection -> fp32 d_out
    gemm_bf16<0><<<dim3(16, 32), 256, 0, stream>>>(attnb, WoutT, out, nullptr, nullptr, nullptr, 2048, 2048);
}
